// CrossAttention_22600117911813
// MI455X (gfx1250) — matrix operand order repaired and checked by reading, not compiled, not run on hardware
//
#include <hip/hip_runtime.h>

typedef __attribute__((ext_vector_type(16))) __bf16 v16bf;
typedef __attribute__((ext_vector_type(8)))  __bf16 v8bf;
typedef __attribute__((ext_vector_type(8)))  float  v8f;

// ---- gfx1250 async global->LDS path (inline asm; falls back to sync loads) ----
#if defined(__gfx1250__)
#define USE_ASYNC_LDS 1
#else
#define USE_ASYNC_LDS 0
#endif

#if USE_ASYNC_LDS
__device__ __forceinline__ void async_cp16(const __bf16* g, __bf16* l) {
    // LDS aperture: flat addr[31:0] is the LDS byte offset (CDNA5 ISA 10.2)
    const unsigned           lds = (unsigned)(unsigned long long)l;
    const unsigned long long ga  = (unsigned long long)g;
    asm volatile("global_load_async_to_lds_b128 %0, %1, off"
                 :: "v"(lds), "v"(ga) : "memory");
}
__device__ __forceinline__ void wait_async0() {
#if __has_builtin(__builtin_amdgcn_s_wait_asynccnt)
    __builtin_amdgcn_s_wait_asynccnt(0);
#else
    asm volatile("s_wait_asynccnt 0x0" ::: "memory");
#endif
}
#endif

// ---------- bf16 helpers (bit-exact RNE, no libcalls) ----------
__device__ __forceinline__ __bf16 f2bf(float f) {
    unsigned u = __builtin_bit_cast(unsigned, f);
    u += 0x7FFFu + ((u >> 16) & 1u);
    unsigned short h = (unsigned short)(u >> 16);
    return __builtin_bit_cast(__bf16, h);
}

// =================================================================
// 1) LayerNorm: e (rows x 1024 f32) -> en (bf16)
// =================================================================
__global__ __launch_bounds__(256) void ln_kernel(
    const float* __restrict__ e, const float* __restrict__ g,
    const float* __restrict__ beta, __bf16* __restrict__ out)
{
    const int row  = blockIdx.x;
    const int tid  = threadIdx.x;
    const int lane = tid & 31;
    const int wv   = tid >> 5;
    const float* x = e + (size_t)row * 1024;

    const float4 x4 = *(const float4*)(x + tid * 4);
    float s  = x4.x + x4.y + x4.z + x4.w;
    float s2 = x4.x*x4.x + x4.y*x4.y + x4.z*x4.z + x4.w*x4.w;
    #pragma unroll
    for (int d = 1; d < 32; d <<= 1) {
        s  += __shfl_xor(s,  d, 32);
        s2 += __shfl_xor(s2, d, 32);
    }
    __shared__ float red[2][8];
    if (lane == 0) { red[0][wv] = s; red[1][wv] = s2; }
    __syncthreads();
    float ts = 0.f, ts2 = 0.f;
    #pragma unroll
    for (int w = 0; w < 8; ++w) { ts += red[0][w]; ts2 += red[1][w]; }
    const float mu   = ts * (1.f / 1024.f);
    const float var  = ts2 * (1.f / 1024.f) - mu * mu;
    const float rstd = rsqrtf(var + 1e-5f);

    const float xv[4] = {x4.x, x4.y, x4.z, x4.w};
    #pragma unroll
    for (int c = 0; c < 4; ++c) {
        const int col = tid * 4 + c;
        const float y = (xv[c] - mu) * rstd * g[col] + beta[col];
        out[(size_t)row * 1024 + col] = f2bf(y);
    }
}

// =================================================================
// 2a) Weight transpose + convert: in (K x N f32) -> out (N x K bf16)
// =================================================================
__global__ __launch_bounds__(256) void transpose_cvt_kernel(
    const float* __restrict__ in, __bf16* __restrict__ out, int K, int N)
{
    __shared__ float tile[32][33];
    const int tx = threadIdx.x & 31;
    const int ty = threadIdx.x >> 5;            // 0..7
    const int n0 = blockIdx.x * 32;
    const int k0 = blockIdx.y * 32;
    #pragma unroll
    for (int rr = 0; rr < 4; ++rr) {
        const int kr = ty + rr * 8;
        tile[kr][tx] = in[(size_t)(k0 + kr) * N + n0 + tx];
    }
    __syncthreads();
    #pragma unroll
    for (int rr = 0; rr < 4; ++rr) {
        const int nr = ty + rr * 8;
        out[(size_t)(n0 + nr) * K + k0 + tx] = f2bf(tile[tx][nr]);
    }
}

// 2b) elementwise f32 -> bf16 (float4 per thread)
__global__ __launch_bounds__(256) void cvt_bf16_kernel(
    const float* __restrict__ in, __bf16* __restrict__ out)
{
    const int i = blockIdx.x * 256 + threadIdx.x;
    const float4 v = ((const float4*)in)[i];
    out[4*i + 0] = f2bf(v.x);
    out[4*i + 1] = f2bf(v.y);
    out[4*i + 2] = f2bf(v.z);
    out[4*i + 3] = f2bf(v.w);
}

// =================================================================
// 3) Tiled GEMM: C[M,N] = A[M,K](bf16) @ Bt[N,K](bf16)^T (+bias)(+res)
//    128x128 block tile, BK=32, 8 waves, wave tile 32x64 (2x4 WMMA).
//    Double-buffered LDS; async global->LDS copies when available.
// =================================================================
template<bool ADD_BIAS, bool ADD_RES, bool OUT_BF16>
__global__ __launch_bounds__(256) void gemm_bf16_kernel(
    const __bf16* __restrict__ A, const __bf16* __restrict__ Bt,
    const float* __restrict__ bias, const float* __restrict__ res,
    void* __restrict__ Cout, int M, int N, int K)
{
    __shared__ __bf16 As[2][128 * 32];
    __shared__ __bf16 Bs[2][128 * 32];

    const int tid  = threadIdx.x;
    const int lane = tid & 31;
    const int wv   = tid >> 5;
    const int hlf  = lane >> 4;     // 0|1: half-wave
    const int l16  = lane & 15;
    const int bm   = blockIdx.y * 128;
    const int bn   = blockIdx.x * 128;
    const int wm   = (wv & 3) * 32; // wave M offset in tile
    const int wn   = (wv >> 2) * 64;// wave N offset in tile

    v8f acc[2][4];
    #pragma unroll
    for (int mi = 0; mi < 2; ++mi)
        #pragma unroll
        for (int ni = 0; ni < 4; ++ni)
            #pragma unroll
            for (int t = 0; t < 8; ++t) acc[mi][ni][t] = 0.f;

    // stage one 128x32 bf16 tile of A and of Bt into buffer `buf`
    auto issue = [&](int k0, int buf) {
        #pragma unroll
        for (int it = 0; it < 2; ++it) {
            const int ch = tid + it * 256;      // 0..511
            const int r  = ch >> 2;             // 0..127
            const int cs = (ch & 3) * 8;        // 0,8,16,24
#if USE_ASYNC_LDS
            async_cp16(A  + (size_t)(bm + r) * K + k0 + cs, &As[buf][r * 32 + cs]);
            async_cp16(Bt + (size_t)(bn + r) * K + k0 + cs, &Bs[buf][r * 32 + cs]);
#else
            *(v8bf*)(&As[buf][r * 32 + cs]) = *(const v8bf*)(A  + (size_t)(bm + r) * K + k0 + cs);
            *(v8bf*)(&Bs[buf][r * 32 + cs]) = *(const v8bf*)(Bt + (size_t)(bn + r) * K + k0 + cs);
#endif
        }
    };

    const int NK = K >> 5;   // K / 32
    issue(0, 0);

    for (int ki = 0; ki < NK; ++ki) {
#if USE_ASYNC_LDS
        wait_async0();
#endif
        __syncthreads();                        // cur buffer ready; prev readers done
        if (ki + 1 < NK) issue((ki + 1) << 5, (ki + 1) & 1);

        const __bf16* Asb = As[ki & 1];
        const __bf16* Bsb = Bs[ki & 1];

        v16bf af[2], bfr[4];
        #pragma unroll
        for (int mi = 0; mi < 2; ++mi) {
            const __bf16* ap = Asb + (wm + mi * 16 + l16) * 32 + hlf * 8;
            const v8bf lo = *(const v8bf*)ap;
            const v8bf hi = *(const v8bf*)(ap + 16);
            #pragma unroll
            for (int t = 0; t < 8; ++t) { af[mi][t] = lo[t]; af[mi][8 + t] = hi[t]; }
        }
        #pragma unroll
        for (int ni = 0; ni < 4; ++ni) {
            const __bf16* bp = Bsb + (wn + ni * 16 + l16) * 32 + hlf * 8;
            const v8bf lo = *(const v8bf*)bp;
            const v8bf hi = *(const v8bf*)(bp + 16);
            #pragma unroll
            for (int t = 0; t < 8; ++t) { bfr[ni][t] = lo[t]; bfr[ni][8 + t] = hi[t]; }
        }
        #pragma unroll
        for (int mi = 0; mi < 2; ++mi)
            #pragma unroll
            for (int ni = 0; ni < 4; ++ni)
                acc[mi][ni] = __builtin_amdgcn_wmma_f32_16x16x32_bf16(
                    false, af[mi], false, bfr[ni], (short)0, acc[mi][ni], false, false);
    }

    #pragma unroll
    for (int mi = 0; mi < 2; ++mi) {
        #pragma unroll
        for (int ni = 0; ni < 4; ++ni) {
            const int col  = bn + wn + ni * 16 + l16;
            const float bv = ADD_BIAS ? bias[col] : 0.f;
            const int row0 = bm + wm + mi * 16 + hlf * 8;
            #pragma unroll
            for (int r = 0; r < 8; ++r) {
                const int row = row0 + r;
                float v = acc[mi][ni][r] + bv;
                if (ADD_RES) v += res[(size_t)row * N + col];
                if (OUT_BF16) ((__bf16*)Cout)[(size_t)row * N + col] = f2bf(v);
                else          ((float*) Cout)[(size_t)row * N + col] = v;
            }
        }
    }
}

// =================================================================
// 4) Fused attention per (bc, n, head): S=QK^T, softmax, O=PV
//    block = 128 threads = 4 waves; each wave: 32 query rows x 64
// =================================================================
__global__ __launch_bounds__(128) void attn_kernel(
    const __bf16* __restrict__ Q, const __bf16* __restrict__ Kb,
    const __bf16* __restrict__ Vb, __bf16* __restrict__ O)
{
    __shared__ __bf16 Vt[64 * 72];      // V^T: [d][j], stride 72
    __shared__ __bf16 Pt[4][32 * 72];   // per-wave P tile [i][j]

    const int blk  = blockIdx.x;
    const int head = blk & 15;
    const int nn   = (blk >> 4) & 1;
    const int bc   = blk >> 5;
    const size_t qrow0 = ((size_t)bc * 2 + nn) * 128;
    const size_t krow0 = (size_t)bc * 64;
    const int col0 = head * 64;

    const int tid = threadIdx.x;

    // stage V^T into LDS (vectorized global read, scattered LDS write)
    #pragma unroll
    for (int it = 0; it < 4; ++it) {
        const int ch = tid + it * 128;      // 0..511
        const int j  = ch >> 3;             // key row 0..63
        const int ds = (ch & 7) * 8;        // d segment
        const v8bf vv = *(const v8bf*)(Vb + (krow0 + j) * 1024 + col0 + ds);
        #pragma unroll
        for (int t = 0; t < 8; ++t) Vt[(ds + t) * 72 + j] = vv[t];
    }
    __syncthreads();

    const int lane = tid & 31;
    const int wv   = tid >> 5;
    const int hlf  = lane >> 4;
    const int l16  = lane & 15;
    const int wm   = wv * 32;

    // ---- S = (Q @ K^T) over d=64 (2 WMMA K-steps) ----
    v8f s[2][4];
    #pragma unroll
    for (int mi = 0; mi < 2; ++mi)
        #pragma unroll
        for (int ni = 0; ni < 4; ++ni)
            #pragma unroll
            for (int t = 0; t < 8; ++t) s[mi][ni][t] = 0.f;

    #pragma unroll
    for (int ks = 0; ks < 2; ++ks) {
        v16bf af[2], bfr[4];
        #pragma unroll
        for (int mi = 0; mi < 2; ++mi) {
            const __bf16* ap = Q + (qrow0 + wm + mi * 16 + l16) * 1024 + col0 + ks * 32 + hlf * 8;
            const v8bf lo = *(const v8bf*)ap;
            const v8bf hi = *(const v8bf*)(ap + 16);
            #pragma unroll
            for (int t = 0; t < 8; ++t) { af[mi][t] = lo[t]; af[mi][8 + t] = hi[t]; }
        }
        #pragma unroll
        for (int ni = 0; ni < 4; ++ni) {
            const __bf16* bp = Kb + (krow0 + ni * 16 + l16) * 1024 + col0 + ks * 32 + hlf * 8;
            const v8bf lo = *(const v8bf*)bp;
            const v8bf hi = *(const v8bf*)(bp + 16);
            #pragma unroll
            for (int t = 0; t < 8; ++t) { bfr[ni][t] = lo[t]; bfr[ni][8 + t] = hi[t]; }
        }
        #pragma unroll
        for (int mi = 0; mi < 2; ++mi)
            #pragma unroll
            for (int ni = 0; ni < 4; ++ni)
                s[mi][ni] = __builtin_amdgcn_wmma_f32_16x16x32_bf16(
                    false, af[mi], false, bfr[ni], (short)0, s[mi][ni], false, false);
    }

    // ---- scaled softmax over j (64 cols = 4 frags x 16 lanes of half-wave) ----
    const float scale = 0.125f;   // 1/sqrt(64)
    #pragma unroll
    for (int mi = 0; mi < 2; ++mi) {
        #pragma unroll
        for (int r = 0; r < 8; ++r) {
            float v0 = s[mi][0][r] * scale, v1 = s[mi][1][r] * scale;
            float v2 = s[mi][2][r] * scale, v3 = s[mi][3][r] * scale;
            float mx = fmaxf(fmaxf(v0, v1), fmaxf(v2, v3));
            #pragma unroll
            for (int d = 1; d < 16; d <<= 1) mx = fmaxf(mx, __shfl_xor(mx, d, 32));
            const float e0 = __expf(v0 - mx), e1 = __expf(v1 - mx);
            const float e2 = __expf(v2 - mx), e3 = __expf(v3 - mx);
            float sum = e0 + e1 + e2 + e3;
            #pragma unroll
            for (int d = 1; d < 16; d <<= 1) sum += __shfl_xor(sum, d, 32);
            const float inv = 1.f / sum;
            s[mi][0][r] = e0 * inv; s[mi][1][r] = e1 * inv;
            s[mi][2][r] = e2 * inv; s[mi][3][r] = e3 * inv;
        }
    }

    // ---- relayout P (C-layout) -> A-layout via wave-private LDS ----
    __bf16* P = Pt[wv];
    #pragma unroll
    for (int mi = 0; mi < 2; ++mi)
        #pragma unroll
        for (int ni = 0; ni < 4; ++ni)
            #pragma unroll
            for (int r = 0; r < 8; ++r)
                P[(mi * 16 + hlf * 8 + r) * 72 + ni * 16 + l16] = f2bf(s[mi][ni][r]);
    // wave-private region: DS ops are in-order within a wave, no barrier needed

    // ---- O = P @ V over j=64 (2 WMMA K-steps) ----
    v8f o[2][4];
    #pragma unroll
    for (int mi = 0; mi < 2; ++mi)
        #pragma unroll
        for (int ni = 0; ni < 4; ++ni)
            #pragma unroll
            for (int t = 0; t < 8; ++t) o[mi][ni][t] = 0.f;

    #pragma unroll
    for (int ks = 0; ks < 2; ++ks) {
        v16bf af[2], bfr[4];
        #pragma unroll
        for (int mi = 0; mi < 2; ++mi) {
            const __bf16* ap = P + (mi * 16 + l16) * 72 + ks * 32 + hlf * 8;
            const v8bf lo = *(const v8bf*)ap;
            const v8bf hi = *(const v8bf*)(ap + 16);
            #pragma unroll
            for (int t = 0; t < 8; ++t) { af[mi][t] = lo[t]; af[mi][8 + t] = hi[t]; }
        }
        #pragma unroll
        for (int ni = 0; ni < 4; ++ni) {
            const __bf16* bp = Vt + (ni * 16 + l16) * 72 + ks * 32 + hlf * 8;
            const v8bf lo = *(const v8bf*)bp;
            const v8bf hi = *(const v8bf*)(bp + 16);
            #pragma unroll
            for (int t = 0; t < 8; ++t) { bfr[ni][t] = lo[t]; bfr[ni][8 + t] = hi[t]; }
        }
        #pragma unroll
        for (int mi = 0; mi < 2; ++mi)
            #pragma unroll
            for (int ni = 0; ni < 4; ++ni)
                o[mi][ni] = __builtin_amdgcn_wmma_f32_16x16x32_bf16(
                    false, af[mi], false, bfr[ni], (short)0, o[mi][ni], false, false);
    }

    #pragma unroll
    for (int mi = 0; mi < 2; ++mi)
        #pragma unroll
        for (int ni = 0; ni < 4; ++ni)
            #pragma unroll
            for (int r = 0; r < 8; ++r) {
                const size_t row = qrow0 + wm + mi * 16 + hlf * 8 + r;
                const int col = col0 + ni * 16 + l16;
                O[row * 1024 + col] = f2bf(o[mi][ni][r]);
            }
}

// =================================================================
// Launch: LN -> weight cvt -> Q/K/V GEMMs -> fused attention -> out GEMM
// =================================================================
extern "C" void kernel_launch(void* const* d_in, const int* in_sizes, int n_in,
                              void* d_out, int out_size, void* d_ws, size_t ws_size,
                              hipStream_t stream)
{
    (void)in_sizes; (void)n_in; (void)out_size; (void)ws_size;
    const float* e    = (const float*)d_in[0];   // (2,32,2,128,1024)
    const float* h    = (const float*)d_in[1];   // (2,32,64,1024)
    const float* wq   = (const float*)d_in[2];
    const float* bq   = (const float*)d_in[3];
    const float* wk   = (const float*)d_in[4];
    const float* bk   = (const float*)d_in[5];
    const float* wv   = (const float*)d_in[6];
    const float* bv   = (const float*)d_in[7];
    const float* wo   = (const float*)d_in[8];
    const float* bo   = (const float*)d_in[9];
    const float* ln_g = (const float*)d_in[10];
    const float* ln_b = (const float*)d_in[11];

    constexpr size_t MB = 1ull << 20;
    char* ws = (char*)d_ws;
    __bf16* wqT = (__bf16*)(ws + 0 * MB);    // 2 MB each
    __bf16* wkT = (__bf16*)(ws + 2 * MB);
    __bf16* wvT = (__bf16*)(ws + 4 * MB);
    __bf16* woT = (__bf16*)(ws + 6 * MB);
    __bf16* hb  = (__bf16*)(ws + 8 * MB);    // 8 MB   (4096 x 1024)
    __bf16* en  = (__bf16*)(ws + 16 * MB);   // 32 MB  (16384 x 1024), reused for O
    __bf16* Qb  = (__bf16*)(ws + 48 * MB);   // 32 MB
    __bf16* Kbf = (__bf16*)(ws + 80 * MB);   // 8 MB
    __bf16* Vbf = (__bf16*)(ws + 88 * MB);   // 8 MB -> total 96 MB

    // 1) LayerNorm e -> en (bf16)
    ln_kernel<<<16384, 256, 0, stream>>>(e, ln_g, ln_b, en);

    // 2) weights -> transposed bf16; h -> bf16
    transpose_cvt_kernel<<<dim3(32, 32), 256, 0, stream>>>(wq, wqT, 1024, 1024);
    transpose_cvt_kernel<<<dim3(32, 32), 256, 0, stream>>>(wk, wkT, 1024, 1024);
    transpose_cvt_kernel<<<dim3(32, 32), 256, 0, stream>>>(wv, wvT, 1024, 1024);
    transpose_cvt_kernel<<<dim3(32, 32), 256, 0, stream>>>(wo, woT, 1024, 1024);
    cvt_bf16_kernel<<<4096, 256, 0, stream>>>(h, hb);

    // 3) projections
    gemm_bf16_kernel<true, false, true><<<dim3(8, 128), 256, 0, stream>>>(
        en, wqT, bq, nullptr, Qb, 16384, 1024, 1024);
    gemm_bf16_kernel<true, false, true><<<dim3(8, 32), 256, 0, stream>>>(
        hb, wkT, bk, nullptr, Kbf, 4096, 1024, 1024);
    gemm_bf16_kernel<true, false, true><<<dim3(8, 32), 256, 0, stream>>>(
        hb, wvT, bv, nullptr, Vbf, 4096, 1024, 1024);

    // 4) fused attention (2048 blocks = 64 bc * 2 n * 16 heads); O reuses en
    attn_kernel<<<2048, 128, 0, stream>>>(Qb, Kbf, Vbf, en);

    // 5) output projection + bias + residual -> fp32 out
    gemm_bf16_kernel<true, true, false><<<dim3(8, 128), 256, 0, stream>>>(
        en, woT, bo, e, d_out, 16384, 1024, 1024);
}
